// MoeBlock_68599217651794
// MI455X (gfx1250) — matrix operand order repaired and checked by reading, not compiled, not run on hardware
//
#include <hip/hip_runtime.h>
#include <hip/hip_bf16.h>
#include <math.h>

typedef __bf16 bf16_t;
typedef __attribute__((ext_vector_type(16))) __bf16 v16bf;
typedef __attribute__((ext_vector_type(8)))  float  v8f;

#define T_TOK 32768
#define DDIM  512
#define NEXP  16
#define TOPK  2
#define MDIM  1024
#define TM    32   // tokens per tile
#define MC    64   // M-chunk width

// LDS pitches (bf16 elements): 16B-aligned, dword-stride gcd(.,64)=4 so 16
// consecutive rows hit 16 distinct 4-bank groups (conflict-free b128 access).
#define XS_LD  536
#define HS_LD  88
#define WT_LD  40

// ---------------- gfx1250 async copy: global -> LDS (ASYNCcnt) ----------------
typedef int v4i __attribute__((vector_size(16)));
typedef __attribute__((address_space(1))) v4i gv4i;  // global
typedef __attribute__((address_space(3))) v4i lv4i;  // LDS

__device__ __forceinline__ void async_b128(const void* gsrc, void* ldst) {
  __builtin_amdgcn_global_load_async_to_lds_b128((gv4i*)gsrc, (lv4i*)ldst, 0, 0);
}
__device__ __forceinline__ void wait_async0() {
  __builtin_amdgcn_s_wait_asynccnt(0);
}

// ---------------- vectorized WMMA fragment loads from LDS ----------------
// A 16x32 bf16 row-major [m][k]: lane 0-15 -> M=lane, halves 0..7=K+0..7,
// 8..15=K+16..23; lanes 16-31 -> K base +8.
__device__ __forceinline__ v16bf lds_load_A(const bf16_t* base, int ld, int lane) {
  int row = lane & 15;
  int kb  = (lane >> 4) << 3;
  const bf16_t* p = base + row * ld + kb;
  union { v16bf v; uint4 q[2]; } u;
  u.q[0] = *(const uint4*)(p);
  u.q[1] = *(const uint4*)(p + 16);
  return u.v;
}

// B 32x16 bf16 stored transposed [n][k]: lane 0-15 -> K=0..7,16..23, lanes 16-31 ->
// K=8..15,24..31, N = lane&15. Two 8-wide K runs per lane -> 2x b128.
__device__ __forceinline__ v16bf lds_load_Bt(const bf16_t* base, int ld, int lane) {
  int col = lane & 15;
  int kb  = (lane >> 4) << 3;
  const bf16_t* p = base + col * ld + kb;
  union { v16bf v; uint4 q[2]; } u;
  u.q[0] = *(const uint4*)(p);
  u.q[1] = *(const uint4*)(p + 16);
  return u.v;
}

// ---------------- small utility kernels ----------------
__global__ void zero_f32_kernel(float* __restrict__ p, int n) {
  int i = blockIdx.x * blockDim.x + threadIdx.x;
  int stride = gridDim.x * blockDim.x;
  for (; i < n; i += stride) p[i] = 0.f;
}

__global__ void zero_i32_kernel(int* __restrict__ p, int n) {
  int i = blockIdx.x * blockDim.x + threadIdx.x;
  if (i < n) p[i] = 0;
}

// Transpose + fp32->bf16: src [E][R][C] f32 -> dst [E][C][R] bf16. block=(32,8).
__global__ void cvt_transpose_kernel(const float* __restrict__ src, bf16_t* __restrict__ dst,
                                     int R, int C) {
  __shared__ float tile[32][33];
  const int e  = blockIdx.z;
  const int c0 = blockIdx.x * 32;
  const int r0 = blockIdx.y * 32;
  const float* s = src + (size_t)e * R * C;
  bf16_t*      d = dst + (size_t)e * R * C;
  const int tx = threadIdx.x;
  const int ty = threadIdx.y;
#pragma unroll
  for (int i = 0; i < 32; i += 8)
    tile[ty + i][tx] = s[(size_t)(r0 + ty + i) * C + c0 + tx];
  __syncthreads();
#pragma unroll
  for (int i = 0; i < 32; i += 8)
    d[(size_t)(c0 + ty + i) * R + r0 + tx] = (bf16_t)tile[tx][ty + i];
}

// ---------------- gating: logits, top-2, softmax, counts ----------------
__global__ void gate_kernel(const float* __restrict__ x, const float* __restrict__ wg,
                            int* __restrict__ routeIdx, float* __restrict__ routeW,
                            int* __restrict__ counts) {
  int wave = threadIdx.x >> 5;
  int lane = threadIdx.x & 31;
  int t = blockIdx.x * 8 + wave;
  if (t >= T_TOK) return;
  float acc[NEXP];
#pragma unroll
  for (int e = 0; e < NEXP; ++e) acc[e] = 0.f;
  const float* xrow = x + (size_t)t * DDIM;
  for (int d = lane; d < DDIM; d += 32) {
    float xv = xrow[d];
    const float4* wrow = (const float4*)(wg + (size_t)d * NEXP);
#pragma unroll
    for (int q = 0; q < 4; ++q) {
      float4 wv = wrow[q];
      acc[q * 4 + 0] += xv * wv.x;
      acc[q * 4 + 1] += xv * wv.y;
      acc[q * 4 + 2] += xv * wv.z;
      acc[q * 4 + 3] += xv * wv.w;
    }
  }
#pragma unroll
  for (int e = 0; e < NEXP; ++e) {
#pragma unroll
    for (int off = 16; off >= 1; off >>= 1)
      acc[e] += __shfl_xor(acc[e], off, 32);
  }
  if (lane == 0) {
    int i0 = 0; float v0 = acc[0];
#pragma unroll
    for (int e = 1; e < NEXP; ++e) if (acc[e] > v0) { v0 = acc[e]; i0 = e; }
    int i1 = (i0 == 0) ? 1 : 0; float v1 = acc[i1];
#pragma unroll
    for (int e = 0; e < NEXP; ++e)
      if (e != i0 && acc[e] > v1) { v1 = acc[e]; i1 = e; }
    float p0 = 1.f / (1.f + __expf(v1 - v0));
    float p1 = 1.f - p0;
    routeIdx[t * 2 + 0] = i0;
    routeIdx[t * 2 + 1] = i1;
    routeW[t * 2 + 0] = p0;
    routeW[t * 2 + 1] = p1;
    atomicAdd(&counts[i0], 1);
    atomicAdd(&counts[i1], 1);
  }
}

__global__ void scan_kernel(const int* __restrict__ counts, int* __restrict__ offsets) {
  if (threadIdx.x == 0 && blockIdx.x == 0) {
    int s = 0;
#pragma unroll
    for (int e = 0; e < NEXP; ++e) { offsets[e] = s; s += counts[e]; }
    offsets[NEXP] = s;
  }
}

__global__ void scatter_kernel(const int* __restrict__ routeIdx, const float* __restrict__ routeW,
                               const int* __restrict__ offsets, int* __restrict__ fill,
                               int* __restrict__ bucketTok, float* __restrict__ bucketW) {
  int t = blockIdx.x * blockDim.x + threadIdx.x;
  if (t >= T_TOK) return;
#pragma unroll
  for (int k = 0; k < TOPK; ++k) {
    int e = routeIdx[t * 2 + k];
    int pos = atomicAdd(&fill[e], 1);
    int idx = offsets[e] + pos;
    bucketTok[idx] = t;
    bucketW[idx] = routeW[t * 2 + k];
  }
}

// ---------------- grouped expert GEMM (async-LDS double-buffered) ----------------
// Weights pre-transposed bf16: w0t,w1t = [E][M][D]; wot = [E][D][M].
__global__ __launch_bounds__(256, 1)
void moe_gemm_kernel(const float* __restrict__ x,
                     const bf16_t* __restrict__ w0t,
                     const bf16_t* __restrict__ w1t,
                     const bf16_t* __restrict__ wot,
                     const int* __restrict__ offsets,
                     const int* __restrict__ bucketTok,
                     const float* __restrict__ bucketW,
                     float* __restrict__ out) {
  __shared__ __align__(16) bf16_t Xs [TM][XS_LD];       // gathered tokens (A, phase A)
  __shared__ __align__(16) bf16_t Hs [TM][HS_LD];       // silu(g0)*g1 chunk (A, phase B)
  __shared__ __align__(16) bf16_t W0sT[2][MC][WT_LD];   // w0 slab, [n][k], ping-pong
  __shared__ __align__(16) bf16_t W1sT[2][MC][WT_LD];   // w1 slab, [n][k], ping-pong
  __shared__ __align__(16) bf16_t WosT[2][DDIM][WT_LD]; // wo slab, [n][k], ping-pong
  __shared__ int    sTok[TM];
  __shared__ float  sW[TM];

  const int e = blockIdx.y;
  const int start = offsets[e];
  const int end = offsets[e + 1];
  const int tile0 = start + blockIdx.x * TM;
  if (tile0 >= end) return;
  const int rows = min(TM, end - tile0);

  const int tid = threadIdx.x;
  const int lane = tid & 31;
  const int wv = tid >> 5;

  if (tid < TM) {
    bool valid = tid < rows;
    sTok[tid] = valid ? bucketTok[tile0 + tid] : 0;
    sW[tid]   = valid ? bucketW[tile0 + tid]   : 0.f;
  }
  __syncthreads();

  { // gather X tile (fp32 -> bf16)
    int r = tid >> 3;
    int c0 = (tid & 7) * 64;
    if (r < rows) {
      const float* xr = x + (size_t)sTok[r] * DDIM + c0;
#pragma unroll
      for (int c = 0; c < 64; c += 4) {
        float4 v = *(const float4*)(xr + c);
        Xs[r][c0 + c + 0] = (bf16_t)v.x;
        Xs[r][c0 + c + 1] = (bf16_t)v.y;
        Xs[r][c0 + c + 2] = (bf16_t)v.z;
        Xs[r][c0 + c + 3] = (bf16_t)v.w;
      }
    } else {
#pragma unroll
      for (int c = 0; c < 64; ++c) Xs[r][c0 + c] = (bf16_t)0.f;
    }
  }
  __syncthreads();

  const bf16_t* w0te = w0t + (size_t)e * DDIM * MDIM;  // [M][D]
  const bf16_t* w1te = w1t + (size_t)e * DDIM * MDIM;  // [M][D]
  const bf16_t* wote = wot + (size_t)e * MDIM * DDIM;  // [D][M]

  const int rt = wv & 1;              // row tile, both phases
  const int ctA = wv >> 1;            // phase-A col tile over MC
  const int cbase = (wv >> 1) * 128;  // phase-B col base over D

  // per-thread staging slots
  const int nA = tid >> 2;            // 0..63
  const int cA = (tid & 3) * 8;       // 0,8,16,24

  v8f outAcc[8];
#pragma unroll
  for (int j = 0; j < 8; ++j) outAcc[j] = {};

  for (int mc = 0; mc < MDIM; mc += MC) {
    // prologue: async-stage k-slab 0 of w0/w1 into buffer 0
    async_b128(w0te + (size_t)(mc + nA) * DDIM + cA, &W0sT[0][nA][cA]);
    async_b128(w1te + (size_t)(mc + nA) * DDIM + cA, &W1sT[0][nA][cA]);

    // ---- phase A: G0 = X w0[:,mc:mc+MC], G1 = X w1[:,mc:mc+MC] ----
    v8f g0 = {}, g1 = {};
    for (int kk = 0; kk < DDIM; kk += 32) {
      const int cur = (kk >> 5) & 1;
      wait_async0();          // own slab landed in LDS
      __syncthreads();        // everyone's slab landed; prior-buffer reads retired
      if (kk + 32 < DDIM) {   // prefetch next slab into the other buffer
        async_b128(w0te + (size_t)(mc + nA) * DDIM + (kk + 32) + cA, &W0sT[cur ^ 1][nA][cA]);
        async_b128(w1te + (size_t)(mc + nA) * DDIM + (kk + 32) + cA, &W1sT[cur ^ 1][nA][cA]);
      }
      v16bf a  = lds_load_A (&Xs[rt * 16][kk], XS_LD, lane);
      v16bf b0 = lds_load_Bt(&W0sT[cur][ctA * 16][0], WT_LD, lane);
      v16bf b1 = lds_load_Bt(&W1sT[cur][ctA * 16][0], WT_LD, lane);
      g0 = __builtin_amdgcn_wmma_f32_16x16x32_bf16(false, a, false, b0, (short)0, g0, false, false);
      g1 = __builtin_amdgcn_wmma_f32_16x16x32_bf16(false, a, false, b1, (short)0, g1, false, false);
    }

    // async-stage wo k-slab 0 into buffer 0 (overlaps SiLU math below)
#pragma unroll
    for (int rr = 0; rr < 2; ++rr) {
      int n = tid * 2 + rr;
      const bf16_t* src = wote + (size_t)n * MDIM + mc;
#pragma unroll
      for (int c = 0; c < 4; ++c)
        async_b128(src + c * 8, &WosT[0][n][c * 8]);
    }

    // ---- H = silu(G0) * G1 -> Hs (bf16 A-operand layout) ----
#pragma unroll
    for (int i = 0; i < 8; ++i) {
      float v0 = g0[i];
      float sig = __builtin_amdgcn_rcpf(1.f + __expf(-v0));
      float h = v0 * sig * g1[i];
      int m = rt * 16 + i + ((lane >> 4) << 3);
      int n = ctA * 16 + (lane & 15);
      Hs[m][n] = (bf16_t)h;
    }

    // ---- phase B: Out += H @ wo[mc:mc+MC, :] ----
    for (int kk2 = 0; kk2 < MC; kk2 += 32) {
      const int cur = (kk2 >> 5) & 1;
      wait_async0();
      __syncthreads();        // wo slab + Hs visible to all
      if (kk2 + 32 < MC) {    // prefetch next wo slab
#pragma unroll
        for (int rr = 0; rr < 2; ++rr) {
          int n = tid * 2 + rr;
          const bf16_t* src = wote + (size_t)n * MDIM + mc + kk2 + 32;
#pragma unroll
          for (int c = 0; c < 4; ++c)
            async_b128(src + c * 8, &WosT[cur ^ 1][n][c * 8]);
        }
      }
      v16bf a = lds_load_A(&Hs[rt * 16][kk2], HS_LD, lane);
#pragma unroll
      for (int j = 0; j < 8; ++j) {
        v16bf b = lds_load_Bt(&WosT[cur][cbase + j * 16][0], WT_LD, lane);
        outAcc[j] = __builtin_amdgcn_wmma_f32_16x16x32_bf16(false, a, false, b, (short)0, outAcc[j], false, false);
      }
    }
  }

  // ---- weighted scatter-add (each token has exactly 2 expert contributions) ----
#pragma unroll
  for (int j = 0; j < 8; ++j) {
    int col = cbase + j * 16 + (lane & 15);
#pragma unroll
    for (int i = 0; i < 8; ++i) {
      int m = rt * 16 + i + ((lane >> 4) << 3);
      if (m < rows) {
        float v = sW[m] * outAcc[j][i];
        unsafeAtomicAdd(&out[(size_t)sTok[m] * DDIM + col], v);
      }
    }
  }
}

// ---------------- host launcher ----------------
extern "C" void kernel_launch(void* const* d_in, const int* in_sizes, int n_in,
                              void* d_out, int out_size, void* d_ws, size_t ws_size,
                              hipStream_t stream) {
  const float* x  = (const float*)d_in[0];
  const float* wg = (const float*)d_in[1];
  const float* w0 = (const float*)d_in[2];
  const float* w1 = (const float*)d_in[3];
  const float* wo = (const float*)d_in[4];
  float* out = (float*)d_out;

  const size_t nW = (size_t)NEXP * DDIM * MDIM;

  char* ws = (char*)d_ws;
  bf16_t* w0t = (bf16_t*)ws; ws += nW * sizeof(bf16_t);  // [E][M][D]
  bf16_t* w1t = (bf16_t*)ws; ws += nW * sizeof(bf16_t);  // [E][M][D]
  bf16_t* wot = (bf16_t*)ws; ws += nW * sizeof(bf16_t);  // [E][D][M]
  int*   routeIdx  = (int*)ws;   ws += (size_t)T_TOK * TOPK * sizeof(int);
  float* routeW    = (float*)ws; ws += (size_t)T_TOK * TOPK * sizeof(float);
  int*   bucketTok = (int*)ws;   ws += (size_t)T_TOK * TOPK * sizeof(int);
  float* bucketW   = (float*)ws; ws += (size_t)T_TOK * TOPK * sizeof(float);
  int*   counts    = (int*)ws;   ws += 64 * sizeof(int);
  int*   fill      = (int*)ws;   ws += 64 * sizeof(int);
  int*   offsets   = (int*)ws;   ws += 64 * sizeof(int);

  zero_i32_kernel<<<1, 128, 0, stream>>>(counts, 128);  // counts + fill
  zero_f32_kernel<<<2048, 256, 0, stream>>>(out, out_size);

  {
    dim3 blk(32, 8);
    dim3 g01(MDIM / 32, DDIM / 32, NEXP);
    cvt_transpose_kernel<<<g01, blk, 0, stream>>>(w0, w0t, DDIM, MDIM);
    cvt_transpose_kernel<<<g01, blk, 0, stream>>>(w1, w1t, DDIM, MDIM);
    dim3 go(DDIM / 32, MDIM / 32, NEXP);
    cvt_transpose_kernel<<<go, blk, 0, stream>>>(wo, wot, MDIM, DDIM);
  }

  gate_kernel<<<T_TOK / 8, 256, 0, stream>>>(x, wg, routeIdx, routeW, counts);
  scan_kernel<<<1, 32, 0, stream>>>(counts, offsets);
  scatter_kernel<<<(T_TOK + 255) / 256, 256, 0, stream>>>(routeIdx, routeW, offsets, fill,
                                                          bucketTok, bucketW);

  dim3 grid(T_TOK / TM, NEXP);
  moe_gemm_kernel<<<grid, 256, 0, stream>>>(x, w0t, w1t, wot, offsets, bucketTok, bucketW, out);
}
